// EfficientAttention_44753559224682
// MI455X (gfx1250) — matrix operand order repaired and checked by reading, not compiled, not run on hardware
//
#include <hip/hip_runtime.h>
#include <hip/hip_bf16.h>
#include <cstdint>
#include <cstddef>

// ---------------------------------------------------------------------------
// EfficientAttention (linear attention) for MI455X / gfx1250, bf16 WMMA path.
//   x:[4,4096,1024] f32 -> out:[4,4096,1024] f32
// Big GEMMs: 128x128 tiles, double-buffered LDS staging via
// global_load_async_to_lds_b128 (ASYNCcnt), v_wmma_f32_16x16x32_bf16 math.
// ---------------------------------------------------------------------------

typedef __attribute__((ext_vector_type(16))) __bf16 v16bf;
typedef __attribute__((ext_vector_type(8)))  __bf16 v8bf;
typedef __attribute__((ext_vector_type(8)))  float  v8f;
typedef __attribute__((ext_vector_type(4)))  int    v4i;

#define WMMA_BF16(A, B, C) \
  __builtin_amdgcn_wmma_f32_16x16x32_bf16(false, (A), false, (B), (short)0, (C), false, false)

static constexpr int BATCH    = 4;
static constexpr int SEQ      = 4096;
static constexpr int DIM      = 1024;
static constexpr int NHEADS   = 16;
static constexpr int HDIM     = 64;
static constexpr int MROWS    = BATCH * SEQ;       // 16384
static constexpr int N_QKV    = 3 * DIM;           // 3072
static constexpr int LDS_LD   = 40;                // padded row stride (bf16 elems), 80 B

// ---- gfx1250 async global->LDS path (guarded; sync fallback otherwise) ----
#define AS1 __attribute__((address_space(1)))
#define AS3 __attribute__((address_space(3)))

#if defined(__has_builtin)
#  if __has_builtin(__builtin_amdgcn_global_load_async_to_lds_b128)
#    define USE_ASYNC_LDS 1
#  endif
#endif
#ifndef USE_ASYNC_LDS
#  define USE_ASYNC_LDS 0
#endif

__device__ __forceinline__ void async_copy16(const __bf16* src, __bf16* lds_dst) {
#if USE_ASYNC_LDS
  __builtin_amdgcn_global_load_async_to_lds_b128(
      (AS1 v4i*)(uintptr_t)src,
      (AS3 v4i*)(uint32_t)(uintptr_t)lds_dst,   // low 32 bits of flat addr = LDS offset
      0, 0);
#else
  *(v8bf*)lds_dst = *(const v8bf*)src;
#endif
}

__device__ __forceinline__ void wait_async_copies() {
#if USE_ASYNC_LDS
#  if defined(__has_builtin) && __has_builtin(__builtin_amdgcn_s_wait_asynccnt)
  __builtin_amdgcn_s_wait_asynccnt(0);
#  else
  asm volatile("s_wait_asynccnt 0x0" ::: "memory");
#  endif
#endif
}

__device__ __forceinline__ v8f vzero8() {
  v8f z;
#pragma unroll
  for (int i = 0; i < 8; ++i) z[i] = 0.0f;
  return z;
}

// A fragment (16x32, M x K), row-major source with leading dim `ld` (elements).
// lane L (0..15): row = L, K = 0..15; lane L (16..31): row = L-16, K = 16..31
__device__ __forceinline__ v16bf load_a_frag(const __bf16* base, int ld, int lane) {
  const int row = lane & 15;
  const int kb  = (lane < 16) ? 0 : 16;
  const __bf16* p = base + (size_t)row * ld + kb;
  v8bf lo = *(const v8bf*)(p);
  v8bf hi = *(const v8bf*)(p + 8);
  v16bf a;
#pragma unroll
  for (int i = 0; i < 8; ++i) { a[i] = lo[i]; a[8 + i] = hi[i]; }
  return a;
}

// B fragment (32x16, K x N) with B[k][n] = W[n][k]; W-style row-major, ld `ldw`.
// lane L: col = L&15; K = (L<16 ? 0..15 : 16..31). Two 16B loads (LDS friendly).
__device__ __forceinline__ v16bf load_b_frag_wT(const __bf16* Wbase, int ldw, int lane) {
  const int col = lane & 15;
  const int ko  = (lane < 16) ? 0 : 16;
  const __bf16* p = Wbase + (size_t)col * ldw + ko;
  v8bf lo = *(const v8bf*)(p);
  v8bf hi = *(const v8bf*)(p + 8);
  v16bf b;
#pragma unroll
  for (int i = 0; i < 8; ++i) { b[i] = lo[i]; b[8 + i] = hi[i]; }
  return b;
}

// Stage a 128-row x 32-col bf16 tile (row ld `ldg`) into LDS rows of stride LDS_LD.
// 256 threads, 2 x 16B chunks each.
__device__ __forceinline__ void stage_tile(const __bf16* __restrict__ g, int ldg, int k0,
                                           __bf16* lds, int tid) {
#pragma unroll
  for (int i = 0; i < 2; ++i) {
    const int idx = tid + 256 * i;        // 0..511
    const int row = idx >> 2;             // 0..127
    const int q   = idx & 3;              // 16B quarter of the 64B row
    async_copy16(g + (size_t)row * ldg + k0 + 8 * q, lds + row * LDS_LD + 8 * q);
  }
}

// Per-K-step compute: wave (wm in 0..3 -> 32 rows, wn in 0..1 -> 64 cols).
__device__ __forceinline__ void compute_step(const __bf16* As, const __bf16* Bs,
                                             int wm, int wn, int lane, v8f acc[2][4]) {
  v16bf a[2];
#pragma unroll
  for (int mi = 0; mi < 2; ++mi)
    a[mi] = load_a_frag(As + (32 * wm + 16 * mi) * LDS_LD, LDS_LD, lane);
#pragma unroll
  for (int j = 0; j < 4; ++j) {
    v16bf b = load_b_frag_wT(Bs + (64 * wn + 16 * j) * LDS_LD, LDS_LD, lane);
#pragma unroll
    for (int mi = 0; mi < 2; ++mi)
      acc[mi][j] = WMMA_BF16(a[mi], b, acc[mi][j]);
  }
}

// ---------------------------------------------------------------------------
// Kernel 0: f32 -> bf16 elementwise convert
// ---------------------------------------------------------------------------
__global__ __launch_bounds__(256) void cvt_f32_bf16(const float* __restrict__ in,
                                                    __bf16* __restrict__ out, int n) {
  int i = blockIdx.x * blockDim.x + threadIdx.x;
  if (i < n) out[i] = (__bf16)in[i];
}

// ---------------------------------------------------------------------------
// Kernel 1: QKV GEMM (qkv = x @ Wqkv^T + b), 128x128 tile, LDS double buffer.
// Fused q-softmax(head_dim) epilogue; outputs routed to q/k/v [b*h][s][64].
// ---------------------------------------------------------------------------
__global__ __launch_bounds__(256) void gemm_qkv(
    const __bf16* __restrict__ x_bf,   // [16384,1024]
    const __bf16* __restrict__ w_bf,   // [3072,1024]
    const float*  __restrict__ b_qkv,  // [3072]
    __bf16* __restrict__ q_bf,         // [64][4096][64]
    __bf16* __restrict__ k_bf,
    __bf16* __restrict__ v_bf) {
  __shared__ alignas(32) __bf16 As[2][128 * LDS_LD];
  __shared__ alignas(32) __bf16 Bs[2][128 * LDS_LD];

  const int n0   = blockIdx.x * 128;
  const int m0   = blockIdx.y * 128;
  const int tid  = threadIdx.x;
  const int w    = tid >> 5;
  const int lane = tid & 31;
  const int wm   = w & 3;    // 4 M-blocks of 32 rows
  const int wn   = w >> 2;   // 2 N-blocks of 64 cols (one head each)

  v8f acc[2][4];
#pragma unroll
  for (int mi = 0; mi < 2; ++mi)
#pragma unroll
    for (int j = 0; j < 4; ++j) acc[mi][j] = vzero8();

  const __bf16* Ag = x_bf + (size_t)m0 * DIM;
  const __bf16* Wg = w_bf + (size_t)n0 * DIM;

  stage_tile(Ag, DIM, 0, As[0], tid);
  stage_tile(Wg, DIM, 0, Bs[0], tid);

  int buf = 0;
  for (int k0 = 0; k0 < DIM; k0 += 32) {
    wait_async_copies();
    __syncthreads();
    if (k0 + 32 < DIM) {
      stage_tile(Ag, DIM, k0 + 32, As[buf ^ 1], tid);
      stage_tile(Wg, DIM, k0 + 32, Bs[buf ^ 1], tid);
    }
    compute_step(As[buf], Bs[buf], wm, wn, lane, acc);
    buf ^= 1;
  }

  const int nc0 = n0 + 64 * wn;        // wave's 64-col (one head) base
  // bias (column-only dependence)
#pragma unroll
  for (int j = 0; j < 4; ++j) {
    const float bv = b_qkv[nc0 + 16 * j + (lane & 15)];
#pragma unroll
    for (int mi = 0; mi < 2; ++mi)
#pragma unroll
      for (int r = 0; r < 8; ++r) acc[mi][j][r] += bv;
  }

  const int section = nc0 >> 10;       // 0=q, 1=k, 2=v
  const int h       = (nc0 >> 6) & 15;
  const int rowadd  = (lane < 16) ? 0 : 8;
  const int coll    = lane & 15;

  if (section == 0) {
    // softmax over the head's 64 columns of each row, * HDIM^-0.5 (=0.125)
#pragma unroll
    for (int mi = 0; mi < 2; ++mi)
#pragma unroll
      for (int r = 0; r < 8; ++r) {
        float vmax = -3.4e38f;
#pragma unroll
        for (int j = 0; j < 4; ++j) vmax = fmaxf(vmax, acc[mi][j][r]);
#pragma unroll
        for (int m = 1; m <= 8; m <<= 1) vmax = fmaxf(vmax, __shfl_xor(vmax, m, 32));
        float e[4], sum = 0.0f;
#pragma unroll
        for (int j = 0; j < 4; ++j) { e[j] = __expf(acc[mi][j][r] - vmax); sum += e[j]; }
#pragma unroll
        for (int m = 1; m <= 8; m <<= 1) sum += __shfl_xor(sum, m, 32);
        const float inv = 0.125f / sum;
        const int M = m0 + 32 * wm + 16 * mi + r + rowadd;
        const int b = M >> 12, s = M & 4095;
        __bf16* dst = q_bf + ((size_t)(b * NHEADS + h) * SEQ + s) * HDIM;
#pragma unroll
        for (int j = 0; j < 4; ++j) dst[16 * j + coll] = (__bf16)(e[j] * inv);
      }
  } else {
    __bf16* outp = (section == 1) ? k_bf : v_bf;
#pragma unroll
    for (int mi = 0; mi < 2; ++mi)
#pragma unroll
      for (int r = 0; r < 8; ++r) {
        const int M = m0 + 32 * wm + 16 * mi + r + rowadd;
        const int b = M >> 12, s = M & 4095;
        __bf16* dst = outp + ((size_t)(b * NHEADS + h) * SEQ + s) * HDIM;
#pragma unroll
        for (int j = 0; j < 4; ++j) dst[16 * j + coll] = (__bf16)acc[mi][j][r];
      }
  }
}

// ---------------------------------------------------------------------------
// Kernel 2: k softmax over seqlen (per (b,h), per column d). In-place on k_bf.
// ---------------------------------------------------------------------------
__global__ __launch_bounds__(256) void k_softmax_seq(__bf16* __restrict__ k_bf) {
  __shared__ float red[4][64];
  __shared__ float m_sh[64];
  __shared__ float d_sh[64];
  const int bh = blockIdx.x;
  const int tid = threadIdx.x;
  const int d = tid & 63, g = tid >> 6;
  __bf16* kp = k_bf + (size_t)bh * SEQ * HDIM;

  float vmax = -3.4e38f;
  for (int s = g; s < SEQ; s += 4) vmax = fmaxf(vmax, (float)kp[(size_t)s * HDIM + d]);
  red[g][d] = vmax;
  __syncthreads();
  if (g == 0) m_sh[d] = fmaxf(fmaxf(red[0][d], red[1][d]), fmaxf(red[2][d], red[3][d]));
  __syncthreads();
  const float m = m_sh[d];

  float sum = 0.0f;
  for (int s = g; s < SEQ; s += 4) sum += __expf((float)kp[(size_t)s * HDIM + d] - m);
  red[g][d] = sum;
  __syncthreads();
  if (g == 0) d_sh[d] = red[0][d] + red[1][d] + red[2][d] + red[3][d];
  __syncthreads();
  const float inv = 1.0f / d_sh[d];

  for (int s = g; s < SEQ; s += 4) {
    const size_t idx = (size_t)s * HDIM + d;
    kp[idx] = (__bf16)(__expf((float)kp[idx] - m) * inv);
  }
}

// ---------------------------------------------------------------------------
// Kernel 3: kv[d][e] = sum_s k[s][d]*v[s][e] per (b,h). M=N=64, K=4096.
// 4 waves split K; per-lane gathers for transposed A; LDS f32 atomic reduce.
// ---------------------------------------------------------------------------
__global__ __launch_bounds__(128) void kv_gemm(const __bf16* __restrict__ k_bf,
                                               const __bf16* __restrict__ v_bf,
                                               float* __restrict__ kv_f) {
  __shared__ float kv_sh[64 * 64];
  const int bh   = blockIdx.x;
  const int tid  = threadIdx.x;
  const int w    = tid >> 5;
  const int lane = tid & 31;

  for (int i = tid; i < 64 * 64; i += 128) kv_sh[i] = 0.0f;
  __syncthreads();

  const __bf16* kb = k_bf + (size_t)bh * SEQ * HDIM;
  const __bf16* vb = v_bf + (size_t)bh * SEQ * HDIM;

  v8f acc[4][4];
#pragma unroll
  for (int mi = 0; mi < 4; ++mi)
#pragma unroll
    for (int j = 0; j < 4; ++j) acc[mi][j] = vzero8();

  const int row_a = lane & 15;
  const int kba   = (lane < 16) ? 0 : 16;
  const int col_b = lane & 15;
  const int kob   = (lane < 16) ? 0 : 16;

  const int kbeg = w * (SEQ / 4), kend = (w + 1) * (SEQ / 4);
  for (int k0 = kbeg; k0 < kend; k0 += 32) {
    v16bf bf[4];
#pragma unroll
    for (int j = 0; j < 4; ++j) {
#pragma unroll
      for (int e = 0; e < 16; ++e)
        bf[j][e] = vb[(size_t)(k0 + kob + e) * HDIM + 16 * j + col_b];
    }
#pragma unroll
    for (int mi = 0; mi < 4; ++mi) {
      v16bf af;
#pragma unroll
      for (int e = 0; e < 8; ++e) {
        af[e]     = kb[(size_t)(k0 + kba + e) * HDIM + 16 * mi + row_a];
        af[8 + e] = kb[(size_t)(k0 + kba + 8 + e) * HDIM + 16 * mi + row_a];
      }
#pragma unroll
      for (int j = 0; j < 4; ++j) acc[mi][j] = WMMA_BF16(af, bf[j], acc[mi][j]);
    }
  }

  const int rowadd = (lane < 16) ? 0 : 8;
#pragma unroll
  for (int mi = 0; mi < 4; ++mi)
#pragma unroll
    for (int j = 0; j < 4; ++j)
#pragma unroll
      for (int r = 0; r < 8; ++r) {
        const int m = 16 * mi + r + rowadd;
        const int n = 16 * j + (lane & 15);
        atomicAdd(&kv_sh[m * 64 + n], acc[mi][j][r]);
      }
  __syncthreads();

  float* outp = kv_f + (size_t)bh * 64 * 64;
  for (int i = tid; i < 64 * 64; i += 128) outp[i] = kv_sh[i];
}

// ---------------------------------------------------------------------------
// Kernel 4: attn = q @ kv per (b,h): M=4096 (tiled 64), N=64, K=64.
// kv staged to LDS as bf16. Output layout [b][s][h*64+d] (= [b,s,dim]).
// ---------------------------------------------------------------------------
__global__ __launch_bounds__(128) void attn_gemm(const __bf16* __restrict__ q_bf,
                                                 const float* __restrict__ kv_f,
                                                 __bf16* __restrict__ attn_bf) {
  __shared__ __bf16 kv_sh[64 * 64];
  const int bh   = blockIdx.y;
  const int s0   = blockIdx.x * 64;
  const int tid  = threadIdx.x;
  const int w    = tid >> 5;
  const int lane = tid & 31;

  const float* kvp = kv_f + (size_t)bh * 64 * 64;
  for (int i = tid; i < 64 * 64; i += 128) kv_sh[i] = (__bf16)kvp[i];
  __syncthreads();

  v8f acc[4];
#pragma unroll
  for (int j = 0; j < 4; ++j) acc[j] = vzero8();

  const __bf16* qrow = q_bf + ((size_t)bh * SEQ + s0 + 16 * w) * HDIM;
  const int col = lane & 15;
  const int ko  = (lane < 16) ? 0 : 16;
#pragma unroll
  for (int k0 = 0; k0 < 64; k0 += 32) {
    v16bf a = load_a_frag(qrow + k0, HDIM, lane);
#pragma unroll
    for (int j = 0; j < 4; ++j) {
      v16bf b;
#pragma unroll
      for (int e = 0; e < 16; ++e) b[e] = kv_sh[(k0 + ko + e) * 64 + 16 * j + col];
      acc[j] = WMMA_BF16(a, b, acc[j]);
    }
  }

  const int bb = bh >> 4, h = bh & 15;
  const int rowadd = (lane < 16) ? 0 : 8;
#pragma unroll
  for (int j = 0; j < 4; ++j)
#pragma unroll
    for (int r = 0; r < 8; ++r) {
      const int s = s0 + 16 * w + r + rowadd;
      const int n = h * HDIM + 16 * j + (lane & 15);
      attn_bf[((size_t)bb * SEQ + s) * DIM + n] = (__bf16)acc[j][r];
    }
}

// ---------------------------------------------------------------------------
// Kernel 5: out = attn @ Wproj^T + b_proj -> f32, 128x128 tile, LDS staging.
// ---------------------------------------------------------------------------
__global__ __launch_bounds__(256) void gemm_proj(const __bf16* __restrict__ A,
                                                 const __bf16* __restrict__ W,
                                                 const float* __restrict__ bias,
                                                 float* __restrict__ out) {
  __shared__ alignas(32) __bf16 As[2][128 * LDS_LD];
  __shared__ alignas(32) __bf16 Bs[2][128 * LDS_LD];

  const int n0   = blockIdx.x * 128;
  const int m0   = blockIdx.y * 128;
  const int tid  = threadIdx.x;
  const int w    = tid >> 5;
  const int lane = tid & 31;
  const int wm   = w & 3;
  const int wn   = w >> 2;

  v8f acc[2][4];
#pragma unroll
  for (int mi = 0; mi < 2; ++mi)
#pragma unroll
    for (int j = 0; j < 4; ++j) acc[mi][j] = vzero8();

  const __bf16* Ag = A + (size_t)m0 * DIM;
  const __bf16* Wg = W + (size_t)n0 * DIM;

  stage_tile(Ag, DIM, 0, As[0], tid);
  stage_tile(Wg, DIM, 0, Bs[0], tid);

  int buf = 0;
  for (int k0 = 0; k0 < DIM; k0 += 32) {
    wait_async_copies();
    __syncthreads();
    if (k0 + 32 < DIM) {
      stage_tile(Ag, DIM, k0 + 32, As[buf ^ 1], tid);
      stage_tile(Wg, DIM, k0 + 32, Bs[buf ^ 1], tid);
    }
    compute_step(As[buf], Bs[buf], wm, wn, lane, acc);
    buf ^= 1;
  }

  const int rowadd = (lane < 16) ? 0 : 8;
#pragma unroll
  for (int j = 0; j < 4; ++j) {
    const int n = n0 + 64 * wn + 16 * j + (lane & 15);
    const float bv = bias[n];
#pragma unroll
    for (int mi = 0; mi < 2; ++mi)
#pragma unroll
      for (int r = 0; r < 8; ++r) {
        const int m = m0 + 32 * wm + 16 * mi + r + rowadd;
        out[(size_t)m * DIM + n] = acc[mi][j][r] + bv;
      }
  }
}

// ---------------------------------------------------------------------------
// Host-side launch
// ---------------------------------------------------------------------------
extern "C" void kernel_launch(void* const* d_in, const int* in_sizes, int n_in,
                              void* d_out, int out_size, void* d_ws, size_t ws_size,
                              hipStream_t stream) {
  const float* x      = (const float*)d_in[0];
  const float* w_qkv  = (const float*)d_in[1];
  const float* b_qkv  = (const float*)d_in[2];
  const float* w_proj = (const float*)d_in[3];
  const float* b_proj = (const float*)d_in[4];
  float* out = (float*)d_out;

  char* ws = (char*)d_ws;
  size_t off = 0;
  auto carve = [&](size_t bytes) {
    void* p = ws + off;
    off += (bytes + 255) & ~(size_t)255;
    return p;
  };
  __bf16* x_bf     = (__bf16*)carve((size_t)MROWS * DIM * 2);
  __bf16* wqkv_bf  = (__bf16*)carve((size_t)N_QKV * DIM * 2);
  __bf16* wproj_bf = (__bf16*)carve((size_t)DIM * DIM * 2);
  __bf16* q_bf     = (__bf16*)carve((size_t)MROWS * DIM * 2);
  __bf16* k_bf     = (__bf16*)carve((size_t)MROWS * DIM * 2);
  __bf16* v_bf     = (__bf16*)carve((size_t)MROWS * DIM * 2);
  float*  kv_f     = (float*)carve((size_t)BATCH * NHEADS * HDIM * HDIM * 4);
  __bf16* attn_bf  = (__bf16*)carve((size_t)MROWS * DIM * 2);
  (void)ws_size; (void)in_sizes; (void)n_in; (void)out_size;

  cvt_f32_bf16<<<(MROWS * DIM + 255) / 256, 256, 0, stream>>>(x, x_bf, MROWS * DIM);
  cvt_f32_bf16<<<(N_QKV * DIM + 255) / 256, 256, 0, stream>>>(w_qkv, wqkv_bf, N_QKV * DIM);
  cvt_f32_bf16<<<(DIM * DIM + 255) / 256, 256, 0, stream>>>(w_proj, wproj_bf, DIM * DIM);

  gemm_qkv<<<dim3(N_QKV / 128, MROWS / 128), 256, 0, stream>>>(
      x_bf, wqkv_bf, b_qkv, q_bf, k_bf, v_bf);

  k_softmax_seq<<<BATCH * NHEADS, 256, 0, stream>>>(k_bf);

  kv_gemm<<<BATCH * NHEADS, 128, 0, stream>>>(k_bf, v_bf, kv_f);

  attn_gemm<<<dim3(SEQ / 64, BATCH * NHEADS), 128, 0, stream>>>(q_bf, kv_f, attn_bf);

  gemm_proj<<<dim3(DIM / 128, MROWS / 128), 256, 0, stream>>>(attn_bf, wproj_bf, b_proj, out);
}
